// MultiHopMessagePassingLayer_43782896615992
// MI455X (gfx1250) — matrix operand order repaired and checked by reading, not compiled, not run on hardware
//
#include <hip/hip_runtime.h>
#include <hip/hip_bf16.h>

typedef __attribute__((ext_vector_type(16))) __bf16 v16bf;
typedef __attribute__((ext_vector_type(8)))  float  v8f;

#define N_NODES   10000
#define NODE_DIM  256
#define EDGE_DIM  16
#define HIDDEN    512
#define N_EDGES   320000
#define K1_REAL   528          // 2*NODE_DIM + EDGE_DIM
#define K1_PAD    544          // padded to multiple of 32 for bf16 WMMA K
#define LDA1      552          // K1_PAD + 8 -> row stride keeps 16B alignment for ds_load_b128
#define LDH       520          // HIDDEN + 8
#define EM        32           // edges per block (M-tile): 2x B-fragment reuse

// ---------- helpers ----------

__device__ __forceinline__ unsigned short f2bf(float f) {
  union { float f; unsigned int u; } v; v.f = f;
  // round-to-nearest-even fp32 -> bf16
  return (unsigned short)((v.u + 0x7FFFu + ((v.u >> 16) & 1u)) >> 16);
}

__device__ __forceinline__ v16bf load_frag_b(const unsigned short* p) {
  union { uint4 q[2]; v16bf v; } u;
  const uint4* p4 = (const uint4*)p;        // fragment-order weights, 16B aligned
  u.q[0] = p4[0];
  u.q[1] = p4[1];
  return u.v;
}

// A-fragment for 16-bit 16x32 WMMA layout: lane (m = lane&15, kg = lane>>4)
// holds K = kt*32 + kg*8 + [0..7]  and  K = kt*32 + 16 + kg*8 + [0..7]
__device__ __forceinline__ v16bf load_frag_a(const unsigned short* row, int kt, int kg) {
  union { uint4 q[2]; v16bf v; } u;
  u.q[0] = *(const uint4*)(row + kt * 32 + kg * 8);
  u.q[1] = *(const uint4*)(row + kt * 32 + 16 + kg * 8);
  return u.v;
}

__device__ __forceinline__ v8f wmma_bf16(v16bf a, v16bf b, v8f c) {
  return __builtin_amdgcn_wmma_f32_16x16x32_bf16(false, a, false, b, (short)0, c,
                                                 false, false);
}

// ---------- preprocessing ----------

// Convert a row-major fp32 (Kreal x Nn) weight matrix into bf16 WMMA
// B-fragments: frag(kTile,nTile) = 32 lanes x 16 bf16 contiguous (1 KB).
// Lane l of a fragment holds column n = nTile*16 + (l&15),
// K = kTile*32 + (l>>4)*8 + h (h<8), kTile*32 + 16 + (l>>4)*8 + (h-8) (h>=8).  K >= Kreal is zero-padded.
__global__ void swizzle_weights(const float* __restrict__ W,
                                unsigned short* __restrict__ Wb,
                                int Kreal, int Kpad, int Nn) {
  const int frag  = blockIdx.x * 8 + (threadIdx.x >> 5);
  const int lane  = threadIdx.x & 31;
  const int nTiles = Nn >> 4;
  const int kTile = frag / nTiles;
  const int nTile = frag - kTile * nTiles;
  if (kTile * 32 >= Kpad) return;
  const int n     = nTile * 16 + (lane & 15);
  const int kBase = kTile * 32 + (lane >> 4) * 8;
  unsigned short* dst = Wb + (long)frag * 512 + lane * 16;
#pragma unroll
  for (int h = 0; h < 16; ++h) {
    const int k = kBase + (h & 7) + ((h >> 3) << 4);
    const float v = (k < Kreal) ? W[(long)k * Nn + n] : 0.0f;
    dst[h] = f2bf(v);
  }
}

__global__ void zero_f32(float* __restrict__ p, long n) {
  const long i = (long)blockIdx.x * blockDim.x + threadIdx.x;
  if (i < n) p[i] = 0.0f;
}

// ---------- fused edge MLP + scatter ----------
// One block = 32 edges (two 16-row M-tiles). 8 waves.
//   GEMM1: (32 x 544) @ (544 x 512) + bias, ReLU  -> sH (bf16)
//   GEMM2: (32 x 512) @ (512 x 256) + bias        -> atomic f32 scatter into agg
// Each B fragment feeds two WMMAs (one per M-tile): halves L2 weight traffic.
__global__ __launch_bounds__(256)
void edge_mlp_kernel(const float* __restrict__ node_f,
                     const int*   __restrict__ eidx,     // [2*E]: targets then sources
                     const float* __restrict__ edge_f,
                     const unsigned short* __restrict__ W1b,
                     const float* __restrict__ bias1,
                     const unsigned short* __restrict__ W2b,
                     const float* __restrict__ bias2,
                     float* __restrict__ agg) {
  __shared__ unsigned short sA[EM * LDA1];
  __shared__ unsigned short sH[EM * LDH];
  __shared__ int sI[EM];

  const int t    = threadIdx.x;
  const int wav  = t >> 5;
  const int lane = t & 31;
  const int nL   = lane & 15;     // N column inside 16x16 C tile (constant per lane)
  const int kg   = lane >> 4;     // K-group (A frag) / M-group (C tile)
  const int mB   = kg * 8;        // C-tile row base for this lane
  const long eBase = (long)blockIdx.x * EM;

  // ---- gather + concat + fp32->bf16 staging of the (32 x 544) A tile ----
  {
    const int m = t >> 3;                   // 8 threads per row, 32 rows
    const long e = eBase + m;
    const int i = eidx[e];
    const int j = eidx[N_EDGES + e];
    if ((t & 7) == 0) sI[m] = i;
    const float* ri = node_f + (long)i * NODE_DIM;
    const float* rj = node_f + (long)j * NODE_DIM;
    const float* re = edge_f + e * EDGE_DIM;
    unsigned short* dst = sA + m * LDA1;
    for (int c = (t & 7); c < K1_PAD; c += 8) {
      float v;
      if      (c < NODE_DIM)     v = ri[c];
      else if (c < 2 * NODE_DIM) v = rj[c - NODE_DIM];
      else if (c < K1_REAL)      v = re[c - 2 * NODE_DIM];
      else                       v = 0.0f;
      dst[c] = f2bf(v);
    }
  }
  __syncthreads();

  // ---- GEMM1: wave owns N-tiles {wav, wav+8, wav+16, wav+24}, both M-tiles ----
  {
    v8f acc[4][2] = {};
    const unsigned short* rowA0 = sA + nL * LDA1;          // M-tile 0, m = lane&15
    const unsigned short* rowA1 = sA + (16 + nL) * LDA1;   // M-tile 1
    for (int kt = 0; kt < K1_PAD / 32; ++kt) {
      const v16bf a0 = load_frag_a(rowA0, kt, kg);
      const v16bf a1 = load_frag_a(rowA1, kt, kg);
      const unsigned short* fb = W1b + ((long)kt * 32 + wav) * 512 + lane * 16;
#pragma unroll
      for (int q = 0; q < 4; ++q) {
        const v16bf b = load_frag_b(fb + (long)q * 8 * 512);
        acc[q][0] = wmma_bf16(a0, b, acc[q][0]);
        acc[q][1] = wmma_bf16(a1, b, acc[q][1]);
      }
    }
#pragma unroll
    for (int q = 0; q < 4; ++q) {
      const int n = (wav + 8 * q) * 16 + nL;
      const float bs = bias1[n];
#pragma unroll
      for (int mt = 0; mt < 2; ++mt) {
#pragma unroll
        for (int r = 0; r < 8; ++r) {
          float v = acc[q][mt][r] + bs;
          v = v > 0.0f ? v : 0.0f;                   // ReLU
          sH[(mt * 16 + mB + r) * LDH + n] = f2bf(v);
        }
      }
    }
  }
  __syncthreads();

  // ---- GEMM2: wave owns N-tiles {wav, wav+8}, both M-tiles; scatter-add ----
  {
    v8f acc[2][2] = {};
    const unsigned short* rowH0 = sH + nL * LDH;
    const unsigned short* rowH1 = sH + (16 + nL) * LDH;
    for (int kt = 0; kt < HIDDEN / 32; ++kt) {
      const v16bf a0 = load_frag_a(rowH0, kt, kg);
      const v16bf a1 = load_frag_a(rowH1, kt, kg);
#pragma unroll
      for (int q = 0; q < 2; ++q) {
        const unsigned short* fb =
            W2b + ((long)kt * 16 + wav + 8 * q) * 512 + lane * 16;
        const v16bf b = load_frag_b(fb);
        acc[q][0] = wmma_bf16(a0, b, acc[q][0]);
        acc[q][1] = wmma_bf16(a1, b, acc[q][1]);
      }
    }
#pragma unroll
    for (int q = 0; q < 2; ++q) {
      const int n = (wav + 8 * q) * 16 + nL;
      const float bs = bias2[n];
#pragma unroll
      for (int mt = 0; mt < 2; ++mt) {
#pragma unroll
        for (int r = 0; r < 8; ++r) {
          atomicAdd(agg + (long)sI[mt * 16 + mB + r] * NODE_DIM + n,
                    acc[q][mt][r] + bs);
        }
      }
    }
  }
}

// ---------- fused node update ----------
// One block = 16 nodes. concat[node_f | agg] -> relu GEMM -> GEMM -> residual.
__global__ __launch_bounds__(256)
void node_update_kernel(const float* __restrict__ node_f,
                        const float* __restrict__ agg,
                        const unsigned short* __restrict__ U1b,
                        const float* __restrict__ bias1,
                        const unsigned short* __restrict__ U2b,
                        const float* __restrict__ bias2,
                        float* __restrict__ out) {
  __shared__ unsigned short sA[16 * LDH];   // 16 x 512 concat input (bf16)
  __shared__ unsigned short sH[16 * LDH];

  const int t    = threadIdx.x;
  const int wav  = t >> 5;
  const int lane = t & 31;
  const int nL   = lane & 15;
  const int kg   = lane >> 4;
  const int mB   = kg * 8;
  const long nBase = (long)blockIdx.x * 16;

  {
    const int m = t >> 4;
    const long node = nBase + m;
    const float* rn = node_f + node * NODE_DIM;
    const float* ra = agg    + node * NODE_DIM;
    unsigned short* dst = sA + m * LDH;
    for (int c = (t & 15); c < HIDDEN; c += 16) {
      const float v = (c < NODE_DIM) ? rn[c] : ra[c - NODE_DIM];
      dst[c] = f2bf(v);
    }
  }
  __syncthreads();

  // GEMM1: (16 x 512) @ (512 x 512), bias + ReLU
  {
    v8f acc[4] = {};
    const unsigned short* rowA = sA + nL * LDH;
    for (int kt = 0; kt < HIDDEN / 32; ++kt) {
      const v16bf a = load_frag_a(rowA, kt, kg);
      const unsigned short* fb = U1b + ((long)kt * 32 + wav) * 512 + lane * 16;
#pragma unroll
      for (int q = 0; q < 4; ++q) {
        const v16bf b = load_frag_b(fb + (long)q * 8 * 512);
        acc[q] = wmma_bf16(a, b, acc[q]);
      }
    }
#pragma unroll
    for (int q = 0; q < 4; ++q) {
      const int n = (wav + 8 * q) * 16 + nL;
      const float bs = bias1[n];
#pragma unroll
      for (int r = 0; r < 8; ++r) {
        float v = acc[q][r] + bs;
        v = v > 0.0f ? v : 0.0f;
        sH[(mB + r) * LDH + n] = f2bf(v);
      }
    }
  }
  __syncthreads();

  // GEMM2: (16 x 512) @ (512 x 256), bias + residual -> out
  {
    v8f acc[2] = {};
    const unsigned short* rowH = sH + nL * LDH;
    for (int kt = 0; kt < HIDDEN / 32; ++kt) {
      const v16bf a = load_frag_a(rowH, kt, kg);
#pragma unroll
      for (int q = 0; q < 2; ++q) {
        const unsigned short* fb =
            U2b + ((long)kt * 16 + wav + 8 * q) * 512 + lane * 16;
        const v16bf b = load_frag_b(fb);
        acc[q] = wmma_bf16(a, b, acc[q]);
      }
    }
#pragma unroll
    for (int q = 0; q < 2; ++q) {
      const int n = (wav + 8 * q) * 16 + nL;
      const float bs = bias2[n];
#pragma unroll
      for (int r = 0; r < 8; ++r) {
        const long node = nBase + mB + r;
        out[node * NODE_DIM + n] =
            node_f[node * NODE_DIM + n] + acc[q][r] + bs;
      }
    }
  }
}

// ---------- launch ----------

extern "C" void kernel_launch(void* const* d_in, const int* in_sizes, int n_in,
                              void* d_out, int out_size, void* d_ws, size_t ws_size,
                              hipStream_t stream) {
  (void)in_sizes; (void)n_in; (void)out_size; (void)ws_size;
  const float* node_f = (const float*)d_in[0];
  const int*   eidx   = (const int*)  d_in[1];
  const float* edge_f = (const float*)d_in[2];
  const float* mW1    = (const float*)d_in[3];
  const float* mb1    = (const float*)d_in[4];
  const float* mW2    = (const float*)d_in[5];
  const float* mb2    = (const float*)d_in[6];
  const float* uW1    = (const float*)d_in[7];
  const float* ub1    = (const float*)d_in[8];
  const float* uW2    = (const float*)d_in[9];
  const float* ub2    = (const float*)d_in[10];
  float* out = (float*)d_out;

  // workspace layout (all offsets multiples of 1 KB)
  char* ws = (char*)d_ws;
  unsigned short* W1b = (unsigned short*)(ws);                 // 544*512*2 = 557056 B
  unsigned short* W2b = (unsigned short*)(ws +  557056);       // 512*256*2 = 262144 B
  unsigned short* U1b = (unsigned short*)(ws +  819200);       // 512*512*2 = 524288 B
  unsigned short* U2b = (unsigned short*)(ws + 1343488);       // 512*256*2 = 262144 B
  float*          agg = (float*)        (ws + 1605632);        // 10000*256*4 bytes

  // bf16 fragment-order weights (re-done every call: deterministic)
  swizzle_weights<<<68, 256, 0, stream>>>(mW1, W1b, K1_REAL, K1_PAD, HIDDEN); // 17*32 frags
  swizzle_weights<<<32, 256, 0, stream>>>(mW2, W2b, HIDDEN, HIDDEN, NODE_DIM);// 16*16 frags
  swizzle_weights<<<64, 256, 0, stream>>>(uW1, U1b, HIDDEN, HIDDEN, HIDDEN);  // 16*32 frags
  swizzle_weights<<<32, 256, 0, stream>>>(uW2, U2b, HIDDEN, HIDDEN, NODE_DIM);// 16*16 frags

  const long aggN = (long)N_NODES * NODE_DIM;
  zero_f32<<<(int)((aggN + 255) / 256), 256, 0, stream>>>(agg, aggN);

  edge_mlp_kernel<<<N_EDGES / EM, 256, 0, stream>>>(
      node_f, eidx, edge_f, W1b, mb1, W2b, mb2, agg);

  node_update_kernel<<<N_NODES / 16, 256, 0, stream>>>(
      node_f, agg, U1b, ub1, U2b, ub2, out);
}
